// RelativeMultiHeadAttention_67345087201604
// MI455X (gfx1250) — matrix operand order repaired and checked by reading, not compiled, not run on hardware
//
#include <hip/hip_runtime.h>

// ---------------------------------------------------------------------------
// CDNA5 (gfx1250) fused relative multi-head attention, f16 WMMA + f32 accum,
// async global->LDS double-buffered GEMM pipeline.
// ---------------------------------------------------------------------------

typedef __attribute__((ext_vector_type(16))) _Float16 v16h;
typedef __attribute__((ext_vector_type(8)))  float    v8f;

#define WMMA_F16(A, B, C) \
  __builtin_amdgcn_wmma_f32_16x16x32_f16(false, (A), false, (B), (short)0, (C), false, false)

__device__ __forceinline__ unsigned short f2h(float f) {
  union { _Float16 h; unsigned short u; } x; x.h = (_Float16)f; return x.u;
}
__device__ __forceinline__ float h2f(unsigned short u) {
  union { unsigned short u; _Float16 h; } x; x.u = u; return (float)x.h;
}

// --- CDNA5 async global->LDS (ASYNCcnt-tracked) ----------------------------
// Generic LDS pointers carry the LDS byte offset in their low 32 bits
// (ISA 10.2: LDS aperture address = addr[31:0]).
__device__ __forceinline__ void async_load_b128(void* lds, const void* gsrc) {
  asm volatile("global_load_async_to_lds_b128 %0, %1, off"
               :: "v"((unsigned)(unsigned long long)lds), "v"(gsrc)
               : "memory");
}
__device__ __forceinline__ void wait_async0() {
  asm volatile("s_wait_asynccnt 0x0" ::: "memory");
}

union FragU { v16h v; uint4 q[2]; };

// A-operand 16x32 f16: lane holds row (lane&15),
// contiguous 16 K values starting at kk + 16*(lane>>4).
__device__ __forceinline__ v16h load_frag_a(const unsigned short* base, int row0, int ld, int kk) {
  const int lane = threadIdx.x & 31;
  const unsigned short* p = base + (row0 + (lane & 15)) * ld + kk + ((lane >> 4) << 4);
  FragU f;
  f.q[0] = *(const uint4*)(p);
  f.q[1] = *(const uint4*)(p + 8);
  return f.v;
}
// B-operand 32x16 f16 (from row-major B^T [n][k]): lane holds col (lane&15),
// contiguous 16 K values starting at kk + 16*(lane>>4).
__device__ __forceinline__ v16h load_frag_b(const unsigned short* base, int col0, int ld, int kk) {
  const int lane = threadIdx.x & 31;
  const unsigned short* p = base + (col0 + (lane & 15)) * ld + kk + ((lane >> 4) << 4);
  FragU f;
  f.q[0] = *(const uint4*)(p);
  f.q[1] = *(const uint4*)(p + 8);
  return f.v;
}

// ---------------------------------------------------------------------------
// Conversions
// ---------------------------------------------------------------------------
__global__ void __launch_bounds__(256) cvt_f32_f16_kernel(const float* __restrict__ in,
                                                          unsigned short* __restrict__ out, int n) {
  int i = blockIdx.x * 256 + threadIdx.x;
  if (i < n) out[i] = f2h(in[i]);
}

// W (1024x1024) f32 -> Wt[n][k] f16, 64x64 LDS-tiled so both sides coalesce.
__global__ void __launch_bounds__(256) transpose_cvt_kernel(const float* __restrict__ W,
                                                            unsigned short* __restrict__ Wt) {
  __shared__ float t[64][65];
  const int tid = threadIdx.x;
  const int bx = blockIdx.x & 15;        // k-tile
  const int by = blockIdx.x >> 4;        // n-tile
  for (int c = tid; c < 4096; c += 256) {
    const int r = c >> 6, cc = c & 63;
    t[r][cc] = W[(size_t)(bx * 64 + r) * 1024 + by * 64 + cc];   // W[k][n]
  }
  __syncthreads();
  for (int c = tid; c < 4096; c += 256) {
    const int r = c >> 6, cc = c & 63;
    Wt[(size_t)(by * 64 + r) * 1024 + bx * 64 + cc] = f2h(t[cc][r]);  // Wt[n][k]
  }
}

// rel_qk (225,1024) f32 -> (240,1024) f16 with zero pad rows 225..239
__global__ void __launch_bounds__(256) cvt_rel_kernel(const float* __restrict__ in,
                                                      unsigned short* __restrict__ out) {
  int i = blockIdx.x * 256 + threadIdx.x;        // 245760 threads
  int row = i >> 10;
  out[i] = (row < 225) ? f2h(in[i]) : (unsigned short)0;
}

// ---------------------------------------------------------------------------
// Tiled GEMM: C[16384 x 1024] = A[16384 x 1024] * Bt^T + bias
// Double-buffered LDS filled by async global->LDS b128 copies.
// MODE 0: q (b,h,s,d) f16 | 1: k (b,h,s,d) f16 | 2: v transposed (b,h,d,s) f16
// MODE 3: out f32 row-major
// ---------------------------------------------------------------------------
__device__ __forceinline__ void gemm_issue_stage(const unsigned short* __restrict__ A,
                                                 const unsigned short* __restrict__ Bt,
                                                 unsigned short* As, unsigned short* Bs,
                                                 int m0, int n0, int kk, int tid) {
  for (int c = tid; c < 512; c += 256) {
    const int r = c >> 2, cc = (c & 3) << 3;
    async_load_b128(As + r * 40 + cc, A + ((size_t)(m0 + r) << 10) + kk + cc);
    async_load_b128(Bs + r * 40 + cc, Bt + ((size_t)(n0 + r) << 10) + kk + cc);
  }
}

template <int MODE>
__global__ void __launch_bounds__(256)
gemm_kernel(const unsigned short* __restrict__ A,    // [16384][1024] f16
            const unsigned short* __restrict__ Bt,   // [1024][1024] f16, Bt[n][k]
            const float* __restrict__ bias,          // [1024]
            void* __restrict__ Out) {
  __shared__ unsigned short As[2][128 * 40];
  __shared__ unsigned short Bs[2][128 * 40];

  const int tid = threadIdx.x;
  const int m0 = blockIdx.y * 128;
  const int n0 = blockIdx.x * 128;
  const int wid = tid >> 5;
  const int wm = wid >> 1;   // 0..3 -> 32-row strip
  const int wn = wid & 1;    // 0..1 -> 64-col strip

  v8f acc[2][4] = {};

  gemm_issue_stage(A, Bt, As[0], Bs[0], m0, n0, 0, tid);
  int p = 0;
  for (int step = 0; step < 32; ++step) {
    wait_async0();             // this wave's stage-p copies landed
    __syncthreads();           // all waves' copies landed; prev compute on p^1 done
    if (step + 1 < 32)
      gemm_issue_stage(A, Bt, As[p ^ 1], Bs[p ^ 1], m0, n0, (step + 1) * 32, tid);

    v16h af[2], bfr[4];
#pragma unroll
    for (int mt = 0; mt < 2; ++mt) af[mt] = load_frag_a(As[p], wm * 32 + mt * 16, 40, 0);
#pragma unroll
    for (int nt = 0; nt < 4; ++nt) bfr[nt] = load_frag_b(Bs[p], wn * 64 + nt * 16, 40, 0);
#pragma unroll
    for (int mt = 0; mt < 2; ++mt)
#pragma unroll
      for (int nt = 0; nt < 4; ++nt)
        acc[mt][nt] = WMMA_F16(af[mt], bfr[nt], acc[mt][nt]);
    p ^= 1;
  }

  const int lane = tid & 31;
  const int half = lane >> 4;
  const int nn = lane & 15;
#pragma unroll
  for (int mt = 0; mt < 2; ++mt)
#pragma unroll
    for (int nt = 0; nt < 4; ++nt)
#pragma unroll
      for (int g = 0; g < 8; ++g) {
        const int m = m0 + wm * 32 + mt * 16 + g + 8 * half;
        const int n = n0 + wn * 64 + nt * 16 + nn;
        const float v = acc[mt][nt][g] + bias[n];
        if (MODE == 3) {
          ((float*)Out)[((size_t)m << 10) + n] = v;
        } else {
          const int b = m >> 6, s = m & 63, h = n >> 6, d = n & 63;
          size_t idx;
          if (MODE == 2) idx = (((size_t)(b * 16 + h)) * 64 + d) * 64 + s;   // v^T per head
          else           idx = (((size_t)(b * 16 + h)) * 64 + s) * 64 + d;   // q/k
          ((unsigned short*)Out)[idx] = f2h(v);
        }
      }
}

// ---------------------------------------------------------------------------
// Fused relative attention per (b,h). 128 threads = 4 waves, wave w owns rows
// [16w, 16w+16). LDS arena (60KB) is time-shared across phases.
// ---------------------------------------------------------------------------
__global__ void __launch_bounds__(128)
relattn_kernel(const unsigned short* __restrict__ qg,      // (b,h,s,d) f16
               const unsigned short* __restrict__ kg,      // (b,h,s,d) f16
               const unsigned short* __restrict__ vtg,     // (b,h,d,s) f16
               const unsigned short* __restrict__ rqk16,   // (240,1024) f16, zero-padded
               const float* __restrict__ relv,             // (15,15,1024) f32
               const float* __restrict__ lbias,            // (16,64,64) f32
               unsigned short* __restrict__ zg) {          // (256,64,1024) f16
  __shared__ __align__(16) unsigned char smem[61440];
  unsigned short* qs    = (unsigned short*)(smem);          // [64][72]  phase A
  unsigned short* ks    = (unsigned short*)(smem + 9216);   // [64][72]  phase A
  unsigned short* Pk    = (unsigned short*)(smem);          // [64][240] phase B
  unsigned short* Pq    = (unsigned short*)(smem + 30720);  // [64][240] phase A/B
  unsigned short* attns = (unsigned short*)(smem);          // [64][72]  phase C
  unsigned short* vts   = (unsigned short*)(smem + 9216);   // [64][72]  phase C
  unsigned short* Aagg  = (unsigned short*)(smem + 18432);  // [64][136] phase C
  unsigned short* rvt   = (unsigned short*)(smem + 35840);  // [64][136] phase C

  const int tid = threadIdx.x;
  const int bh = blockIdx.x;
  const int h = bh & 15;
  const int b = bh >> 4;
  const size_t tbase = (size_t)bh * 4096;

  // stage q, k tiles (contiguous 64x64 f16) -> stride-72 LDS via async copies
  for (int c = tid; c < 512; c += 128) {
    const int s = c >> 3, j = (c & 7) << 3;        // 16B chunks
    async_load_b128(qs + s * 72 + j, qg + tbase + s * 64 + j);
    async_load_b128(ks + s * 72 + j, kg + tbase + s * 64 + j);
  }
  wait_async0();
  __syncthreads();

  const int w = tid >> 5;
  const int lane = tid & 31;
  const int half = lane >> 4;
  const int nn = lane & 15;
  const int r0 = w * 16;
  const unsigned short* rqk_h = rqk16 + h * 64;   // rel table B-frags straight from L2

  v8f accL[4] = {};
  v8f accP[15] = {};
  const v8f vzero = {0.f, 0.f, 0.f, 0.f, 0.f, 0.f, 0.f, 0.f};

  // Phase A1: logits = q k^T and Pq[i][rc] = q_i . rel_qk[rc]
#pragma unroll
  for (int kk = 0; kk < 64; kk += 32) {
    v16h aq = load_frag_a(qs, r0, 72, kk);
#pragma unroll
    for (int jt = 0; jt < 4; ++jt)
      accL[jt] = WMMA_F16(aq, load_frag_b(ks, jt * 16, 72, kk), accL[jt]);
#pragma unroll
    for (int t = 0; t < 15; ++t)
      accP[t] = WMMA_F16(aq, load_frag_b(rqk_h, t * 16, 1024, kk), accP[t]);
  }
#pragma unroll
  for (int t = 0; t < 15; ++t)
#pragma unroll
    for (int g = 0; g < 8; ++g)
      Pq[(r0 + g + 8 * half) * 240 + t * 16 + nn] = f2h(accP[t][g]);

  // Phase A2: Pk[j][rc] = k_j . rel_qk[rc]
#pragma unroll
  for (int t = 0; t < 15; ++t) accP[t] = vzero;
#pragma unroll
  for (int kk = 0; kk < 64; kk += 32) {
    v16h ak = load_frag_a(ks, r0, 72, kk);
#pragma unroll
    for (int t = 0; t < 15; ++t)
      accP[t] = WMMA_F16(ak, load_frag_b(rqk_h, t * 16, 1024, kk), accP[t]);
  }
  __syncthreads();   // everyone done reading qs/ks; Pk overwrites them
#pragma unroll
  for (int t = 0; t < 15; ++t)
#pragma unroll
    for (int g = 0; g < 8; ++g)
      Pk[(r0 + g + 8 * half) * 240 + t * 16 + nn] = f2h(accP[t][g]);
  __syncthreads();   // Pk visible to all waves

  // Phase B: assemble logits (gather rel terms by distance index) + softmax
#pragma unroll
  for (int jt = 0; jt < 4; ++jt)
#pragma unroll
    for (int g = 0; g < 8; ++g) {
      const int i = r0 + g + 8 * half;
      const int j = jt * 16 + nn;
      const int rc = ((j >> 3) - (i >> 3) + 7) * 15 + ((j & 7) - (i & 7) + 7);
      accL[jt][g] = 0.125f * (accL[jt][g]
                              + h2f(Pq[i * 240 + rc])
                              + h2f(Pk[j * 240 + rc])
                              + lbias[((h * 64 + i) << 6) + j]);
    }
#pragma unroll
  for (int g = 0; g < 8; ++g) {
    float m = fmaxf(fmaxf(accL[0][g], accL[1][g]), fmaxf(accL[2][g], accL[3][g]));
#pragma unroll
    for (int mk = 1; mk < 16; mk <<= 1) m = fmaxf(m, __shfl_xor(m, mk, 32));
    float ssum = 0.f;
#pragma unroll
    for (int jt = 0; jt < 4; ++jt) { float e = __expf(accL[jt][g] - m); accL[jt][g] = e; ssum += e; }
#pragma unroll
    for (int mk = 1; mk < 16; mk <<= 1) ssum += __shfl_xor(ssum, mk, 32);
    const float inv = 1.f / ssum;
#pragma unroll
    for (int jt = 0; jt < 4; ++jt) accL[jt][g] *= inv;
  }
  __syncthreads();   // all Pq/Pk reads complete; reuse arena

  // Phase C staging: v^T tile (async) + attn (f16) strip
  for (int c = tid; c < 512; c += 128) {
    const int d = c >> 3, j = (c & 7) << 3;
    async_load_b128(vts + d * 72 + j, vtg + tbase + d * 64 + j);
  }
#pragma unroll
  for (int jt = 0; jt < 4; ++jt)
#pragma unroll
    for (int g = 0; g < 8; ++g)
      attns[(r0 + g + 8 * half) * 72 + jt * 16 + nn] = f2h(accL[jt][g]);
  wait_async0();
  __syncthreads();

  v8f accZ[4] = {};
  // z = attn @ v
#pragma unroll
  for (int kk = 0; kk < 64; kk += 32) {
    v16h aa = load_frag_a(attns, r0, 72, kk);
#pragma unroll
    for (int dt = 0; dt < 4; ++dt)
      accZ[dt] = WMMA_F16(aa, load_frag_b(vts, dt * 16, 72, kk), accZ[dt]);
  }

  // z += (attn scattered into 225 distance bins) @ rel_v, two 128-wide halves
  for (int hh = 0; hh < 2; ++hh) {
    for (int c = tid; c < 64 * 128; c += 128) {   // Aagg[i][rcl]
      const int i = c >> 7, rcl = c & 127;
      const int rc = hh * 128 + rcl;
      const int dr = rc / 15, dc = rc - dr * 15;
      const int grj = (i >> 3) + dr - 7, gcj = (i & 7) + dc - 7;
      unsigned short vv = 0;
      if (dr < 15 && grj >= 0 && grj < 8 && gcj >= 0 && gcj < 8)
        vv = attns[i * 72 + grj * 8 + gcj];
      Aagg[i * 136 + rcl] = vv;
    }
    for (int c = tid; c < 64 * 128; c += 128) {   // rvt[d][rcl] = rel_v[rc][h*64+d]
      const int rcl = c >> 6, d = c & 63;
      const int rc = hh * 128 + rcl;
      rvt[d * 136 + rcl] = (rc < 225) ? f2h(relv[((size_t)rc << 10) + h * 64 + d])
                                      : (unsigned short)0;
    }
    __syncthreads();
#pragma unroll
    for (int kk = 0; kk < 128; kk += 32) {
      v16h aa = load_frag_a(Aagg, r0, 136, kk);
#pragma unroll
      for (int dt = 0; dt < 4; ++dt)
        accZ[dt] = WMMA_F16(aa, load_frag_b(rvt, dt * 16, 136, kk), accZ[dt]);
    }
    __syncthreads();
  }

  // write z (f16) to (b, s, e) row-major for the output projection
#pragma unroll
  for (int dt = 0; dt < 4; ++dt)
#pragma unroll
    for (int g = 0; g < 8; ++g) {
      const int i = r0 + g + 8 * half;
      const int d = dt * 16 + nn;
      zg[((size_t)(b * 64 + i) << 10) + h * 64 + d] = f2h(accZ[dt][g]);
    }
}

// ---------------------------------------------------------------------------
// Host launcher
// ---------------------------------------------------------------------------
extern "C" void kernel_launch(void* const* d_in, const int* in_sizes, int n_in,
                              void* d_out, int out_size, void* d_ws, size_t ws_size,
                              hipStream_t stream) {
  (void)in_sizes; (void)n_in; (void)out_size; (void)ws_size;
  const float* x     = (const float*)d_in[0];
  const float* Wq    = (const float*)d_in[1];
  const float* bq    = (const float*)d_in[2];
  const float* Wk    = (const float*)d_in[3];
  const float* bk    = (const float*)d_in[4];
  const float* Wv    = (const float*)d_in[5];
  const float* bv    = (const float*)d_in[6];
  const float* Wo    = (const float*)d_in[7];
  const float* bo    = (const float*)d_in[8];
  const float* relqk = (const float*)d_in[9];
  const float* relv  = (const float*)d_in[10];
  const float* lbias = (const float*)d_in[11];
  float* out = (float*)d_out;

  unsigned char* ws = (unsigned char*)d_ws;
  const size_t SZ = 33554432ull;                              // 16.7M f16 bytes
  unsigned short* x_h  = (unsigned short*)(ws);               // x f16; reused as z f16
  unsigned short* q_h  = (unsigned short*)(ws + SZ);          // (b,h,s,d)
  unsigned short* k_h  = (unsigned short*)(ws + 2 * SZ);      // (b,h,s,d)
  unsigned short* vt_h = (unsigned short*)(ws + 3 * SZ);      // (b,h,d,s)
  unsigned short* wtq  = (unsigned short*)(ws + 4 * SZ);      // 1024x1024 each
  unsigned short* wtk  = wtq + 1048576;
  unsigned short* wtv  = wtk + 1048576;
  unsigned short* wto  = wtv + 1048576;
  unsigned short* rqkh = wto + 1048576;                       // 240x1024 f16

  cvt_f32_f16_kernel<<<65536, 256, 0, stream>>>(x, x_h, 16777216);
  transpose_cvt_kernel<<<256, 256, 0, stream>>>(Wq, wtq);
  transpose_cvt_kernel<<<256, 256, 0, stream>>>(Wk, wtk);
  transpose_cvt_kernel<<<256, 256, 0, stream>>>(Wv, wtv);
  transpose_cvt_kernel<<<256, 256, 0, stream>>>(Wo, wto);
  cvt_rel_kernel<<<960, 256, 0, stream>>>(relqk, rqkh);

  dim3 gdim(8, 128, 1);
  gemm_kernel<0><<<gdim, 256, 0, stream>>>(x_h, wtq, bq, (void*)q_h);
  gemm_kernel<1><<<gdim, 256, 0, stream>>>(x_h, wtk, bk, (void*)k_h);
  gemm_kernel<2><<<gdim, 256, 0, stream>>>(x_h, wtv, bv, (void*)vt_h);

  relattn_kernel<<<4096, 128, 0, stream>>>(q_h, k_h, vt_h, rqkh, relv, lbias, x_h);

  gemm_kernel<3><<<gdim, 256, 0, stream>>>(x_h, wto, bo, (void*)out);
}
